// AdaptiveFrequencyFilter_69561290326543
// MI455X (gfx1250) — compile-verified, matrix operand order checked by reading, not run on hardware
//
#include <hip/hip_runtime.h>
#include <hip/hip_bf16.h>
#include <math.h>

typedef __attribute__((ext_vector_type(16))) __bf16 v16bf;
typedef __attribute__((ext_vector_type(8)))  float  v8f;

#define B_   16
#define C_   256
#define H_   128
#define W_   128
#define NF_  8
#define HW_  (H_*W_)
#define GROUPS 8
#define GSIZE  32
#define EPS_ 1e-5f

// ---- fp32 -> bf16 round-to-nearest-even ----
__device__ __forceinline__ unsigned short f2bf(float f) {
    union { float f; unsigned u; } v; v.f = f;
    unsigned r = v.u + 0x7fffu + ((v.u >> 16) & 1u);
    return (unsigned short)(r >> 16);
}

// =====================================================================
// Kernel 1: global average pool -> pooled[B,C]
// =====================================================================
__global__ __launch_bounds__(256)
void pool_kernel(const float* __restrict__ x, float* __restrict__ pooled) {
    int bc  = blockIdx.x;            // b*C + c
    int tid = threadIdx.x;
    const float4* p = (const float4*)(x + (size_t)bc * HW_);
    float s = 0.f;
#pragma unroll
    for (int j = 0; j < 16; ++j) {
        float4 v = p[tid + j * 256];
        s += v.x + v.y + v.z + v.w;
    }
#pragma unroll
    for (int off = 16; off > 0; off >>= 1) s += __shfl_xor(s, off, 32);
    __shared__ float red[8];
    if ((tid & 31) == 0) red[tid >> 5] = s;
    __syncthreads();
    if (tid == 0) {
        float t = 0.f;
        for (int i = 0; i < 8; ++i) t += red[i];
        pooled[bc] = t * (1.f / HW_);
    }
}

// =====================================================================
// Kernel 2: selector (GAP @ sel_w.T + b -> softmax) -> combined 7x7/sample
//           also zeroes the GN stats accumulators (256 floats)
// =====================================================================
__global__ __launch_bounds__(256)
void prep_kernel(const float* __restrict__ pooled,
                 const float* __restrict__ sel_w,
                 const float* __restrict__ sel_b,
                 const float* __restrict__ fbank,
                 float* __restrict__ combined,
                 float* __restrict__ stats) {
    int tid = threadIdx.x;
    stats[tid] = 0.f;                 // 16*8*2 = 256 entries
    if (tid < B_) {
        float logit[NF_];
#pragma unroll
        for (int n = 0; n < NF_; ++n) {
            float a = sel_b[n];
            for (int c = 0; c < C_; ++c)
                a = fmaf(pooled[tid * C_ + c], sel_w[n * C_ + c], a);
            logit[n] = a;
        }
        float mx = logit[0];
#pragma unroll
        for (int n = 1; n < NF_; ++n) mx = fmaxf(mx, logit[n]);
        float den = 0.f;
#pragma unroll
        for (int n = 0; n < NF_; ++n) { logit[n] = __expf(logit[n] - mx); den += logit[n]; }
        float inv = 1.f / den;
        for (int j = 0; j < 49; ++j) {
            float v = 0.f;
#pragma unroll
            for (int n = 0; n < NF_; ++n) v = fmaf(logit[n] * inv, fbank[n * 49 + j], v);
            combined[tid * 49 + j] = v;
        }
    }
}

// =====================================================================
// Kernel 3: pack proj_w (fp32 [O=256, I=256]) into bf16 WMMA A-operand
// order.  Tile = [16 M x 32 K]; tiles indexed [mTile][kChunk].
// Per-lane layout (16-bit A 16x32): M = lane%16,
// K = (e>=8)*16 + (lane>=16)*8 + (e&7)   where e = element 0..15.
// =====================================================================
__global__ __launch_bounds__(256)
void pack_kernel(const float* __restrict__ proj_w, unsigned short* __restrict__ Ppk) {
    int f       = blockIdx.x * 256 + threadIdx.x;   // 0..65535
    int tileIdx = f >> 9;
    int sidx    = f & 511;
    int l       = sidx >> 4;
    int e       = sidx & 15;
    int mT      = tileIdx >> 3;
    int kc      = tileIdx & 7;
    int M       = mT * 16 + (l & 15);
    int K       = kc * 32 + ((e >> 3) << 4) + ((l >> 4) << 3) + (e & 7);
    Ppk[f] = f2bf(proj_w[M * C_ + K]);
}

// =====================================================================
// Kernel 4 (core): per (sample b, image row) workgroup.
//   K-loop over 8 chunks of 32 input channels:
//     stage A: 7x7 depthwise conv of the chunk for this row (all 128 cols)
//              -> bf16 LDS tile pre-swizzled into WMMA B-operand order
//     stage B: v_wmma_f32_16x16x32_bf16, A = packed proj_w from global
//   Epilogue: +proj_b, store y, per-group sum/sumsq -> atomics (wave==group)
// =====================================================================
__global__ __launch_bounds__(256)
void conv_gemm_kernel(const float* __restrict__ x,
                      const float* __restrict__ combined,
                      const unsigned short* __restrict__ Ppk,
                      const float* __restrict__ proj_b,
                      float* __restrict__ y,
                      float* __restrict__ stats) {
    const int row = blockIdx.x;
    const int b   = blockIdx.y;
    const int tid = threadIdx.x;
    const int ln  = tid & 31;
    const int wv  = tid >> 5;            // wave id == GN group id (M = 32*wv..32*wv+31)

    __shared__ float filt[49];
    __shared__ float pbias[C_];
    __shared__ unsigned short sB[32 * 128];   // 8 B-tiles of 512 bf16, wmma-swizzled

    if (tid < 49) filt[tid] = combined[b * 49 + tid];
    pbias[tid] = proj_b[tid];
    __syncthreads();

    v8f acc[2][8] = {};

    for (int kc = 0; kc < 8; ++kc) {
        // -------- stage A: depthwise conv (32 ch x 128 cols, this row) --------
#pragma unroll
        for (int it = 0; it < 4; ++it) {
            int item = tid + it * 256;        // 0..1023
            int kl   = item >> 5;             // local channel 0..31
            int n0   = (item & 31) << 2;      // 4 consecutive output cols
            int k    = kc * 32 + kl;
            const float* xp = x + ((size_t)b * C_ + k) * (size_t)HW_;
            // prefetch next chunk's row of this channel slot into GL2
            if (kc < 7)
                __builtin_prefetch(xp + (size_t)32 * HW_ + row * W_ + n0, 0, 1);
            float o0 = 0.f, o1 = 0.f, o2 = 0.f, o3 = 0.f;
#pragma unroll
            for (int dy = 0; dy < 7; ++dy) {
                int ry = row + dy - 3;
                if (ry < 0 || ry >= H_) continue;
                const float* rp = xp + ry * W_;
                float win[10];
#pragma unroll
                for (int t = 0; t < 10; ++t) {
                    int col = n0 - 3 + t;
                    win[t] = (col >= 0 && col < W_) ? rp[col] : 0.f;
                }
#pragma unroll
                for (int dx = 0; dx < 7; ++dx) {
                    float fv = filt[dy * 7 + dx];
                    o0 = fmaf(fv, win[dx + 0], o0);
                    o1 = fmaf(fv, win[dx + 1], o1);
                    o2 = fmaf(fv, win[dx + 2], o2);
                    o3 = fmaf(fv, win[dx + 3], o3);
                }
            }
            // scatter into B-operand order: lane = (n&15) + ((K>>3)&1)*16,
            // elem = (K>=16)*8 + (K&7), tile = n>>4
            int e     = ((kl >> 4) << 3) | (kl & 7);
            int lhalf = ((kl >> 3) & 1) << 4;
            float oo[4] = { o0, o1, o2, o3 };
#pragma unroll
            for (int j = 0; j < 4; ++j) {
                int n   = n0 + j;
                int off = ((n >> 4) << 9) + (((n & 15) + lhalf) << 4) + e;
                sB[off] = f2bf(oo[j]);
            }
        }
        __syncthreads();

        // -------- stage B: WMMA over this K-chunk --------
#pragma unroll
        for (int m = 0; m < 2; ++m) {
            int mT = wv * 2 + m;
            const v16bf A = *(const v16bf*)(Ppk + ((size_t)(mT * 8 + kc)) * 512 + ln * 16);
#pragma unroll
            for (int nT = 0; nT < 8; ++nT) {
                const v16bf Bm = *(const v16bf*)(&sB[(nT << 9) + ln * 16]);
                acc[m][nT] = __builtin_amdgcn_wmma_f32_16x16x32_bf16(
                    false, A, false, Bm, (short)0, acc[m][nT], false, false);
            }
        }
        __syncthreads();
    }

    // -------- epilogue: bias, store y, GN partial stats --------
    float s = 0.f, sq = 0.f;
#pragma unroll
    for (int m = 0; m < 2; ++m) {
        int mT    = wv * 2 + m;
        int mbase = mT * 16 + ((ln >> 4) << 3);
#pragma unroll
        for (int nT = 0; nT < 8; ++nT) {
            int ncol = nT * 16 + (ln & 15);
#pragma unroll
            for (int i = 0; i < 8; ++i) {
                int M   = mbase + i;
                float v = acc[m][nT][i] + pbias[M];
                y[(((size_t)b * C_ + M) * H_ + row) * W_ + ncol] = v;
                s  += v;
                sq += v * v;
            }
        }
    }
#pragma unroll
    for (int off = 16; off > 0; off >>= 1) {
        s  += __shfl_xor(s,  off, 32);
        sq += __shfl_xor(sq, off, 32);
    }
    if (ln == 0) {
        atomicAdd(&stats[(b * GROUPS + wv) * 2 + 0], s);
        atomicAdd(&stats[(b * GROUPS + wv) * 2 + 1], sq);
    }
}

// =====================================================================
// Kernel 5: GroupNorm + exact GELU + residual, in place on d_out
// =====================================================================
__global__ __launch_bounds__(256)
void gn_gelu_kernel(const float* __restrict__ x,
                    const float* __restrict__ stats,
                    const float* __restrict__ gamma,
                    const float* __restrict__ beta,
                    float* __restrict__ y) {
    size_t i4   = (size_t)blockIdx.x * 256 + threadIdx.x;
    size_t base = i4 * 4;
    int b = (int)(base >> 22);            // C*H*W = 2^22
    int c = (int)((base >> 14) & 255);    // H*W   = 2^14
    int g = c >> 5;
    float s    = stats[(b * GROUPS + g) * 2 + 0];
    float sq   = stats[(b * GROUPS + g) * 2 + 1];
    const float invN = 1.f / (float)(GSIZE * HW_);
    float mean = s * invN;
    float var  = fmaxf(sq * invN - mean * mean, 0.f);
    float rs   = rsqrtf(var + EPS_);
    float ga   = gamma[c] * rs;
    float be   = beta[c];

    float4 yv = ((const float4*)y)[i4];
    float4 xv = ((const float4*)x)[i4];
    float r[4]  = { yv.x, yv.y, yv.z, yv.w };
    float xr[4] = { xv.x, xv.y, xv.z, xv.w };
#pragma unroll
    for (int j = 0; j < 4; ++j) {
        float z  = (r[j] - mean) * ga + be;
        float gl = 0.5f * z * (1.f + erff(z * 0.70710678118654752f));
        r[j] = gl + xr[j];
    }
    ((float4*)y)[i4] = make_float4(r[0], r[1], r[2], r[3]);
}

// =====================================================================
// Host launcher
// =====================================================================
extern "C" void kernel_launch(void* const* d_in, const int* in_sizes, int n_in,
                              void* d_out, int out_size, void* d_ws, size_t ws_size,
                              hipStream_t stream) {
    const float* x      = (const float*)d_in[0];
    const float* fbank  = (const float*)d_in[1];   // [8,1,7,7]
    const float* sel_w  = (const float*)d_in[2];   // [8,256]
    const float* sel_b  = (const float*)d_in[3];   // [8]
    const float* proj_w = (const float*)d_in[4];   // [256,256,1,1]
    const float* proj_b = (const float*)d_in[5];   // [256]
    const float* gamma  = (const float*)d_in[6];   // [256]
    const float* beta   = (const float*)d_in[7];   // [256]
    float* out = (float*)d_out;

    char* ws = (char*)d_ws;
    float*          pooled   = (float*)(ws + 0);        // 4096 f32 (16 KB)
    float*          combined = (float*)(ws + 16384);    // 16*49 f32
    float*          stats    = (float*)(ws + 20480);    // 256 f32
    unsigned short* Ppk      = (unsigned short*)(ws + 24576); // 65536 bf16 (128 KB)

    pool_kernel<<<dim3(B_ * C_), 256, 0, stream>>>(x, pooled);
    prep_kernel<<<dim3(1), 256, 0, stream>>>(pooled, sel_w, sel_b, fbank, combined, stats);
    pack_kernel<<<dim3(256), 256, 0, stream>>>(proj_w, Ppk);
    conv_gemm_kernel<<<dim3(H_, B_), 256, 0, stream>>>(x, combined, Ppk, proj_b, out, stats);
    gn_gelu_kernel<<<dim3((B_ * C_ * HW_) / 1024), 256, 0, stream>>>(x, stats, gamma, beta, out);
}
